// DecomposedSpatialTemporalAttention_2413771620433
// MI455X (gfx1250) — compile-verified, matrix operand order checked by reading, not run on hardware
//
#include <hip/hip_runtime.h>

// ---------------------------------------------------------------------------
// DecomposedSpatialTemporalAttention for MI455X (gfx1250, wave32, WMMA bf16)
//  - all five linear layers on v_wmma_f32_16x16x32_bf16 (f32 accumulate)
//  - spatial attention: flash-style, K/V tiles staged with
//    global_load_async_to_lds_b128 (ASYNCcnt) and consumed via ds_load_b128
//  - temporal attention (T=17): VALU + LDS (too small for 16x16x32 tiles)
// ---------------------------------------------------------------------------

typedef __attribute__((ext_vector_type(16))) __bf16  v16bf;
typedef __attribute__((ext_vector_type(8)))  float   v8f;
typedef __attribute__((ext_vector_type(4)))  unsigned int u32x4;

union Frag {
    v16bf v;
    u32x4 q[2];
    __bf16 e[16];
};

static constexpr int B_ = 8, T_ = 17, N_ = 197, C_ = 768, H_ = 12, D_ = 64;
static constexpr int TOK = B_ * T_ * N_;            // 26792 tokens
static constexpr float SCL = 0.125f;                // D^-0.5

__device__ __forceinline__ v8f vzero8() {
    v8f z = {0.f, 0.f, 0.f, 0.f, 0.f, 0.f, 0.f, 0.f};
    return z;
}

// ---------------------------------------------------------------------------
// fp32 -> bf16 weight conversion
// ---------------------------------------------------------------------------
__global__ __launch_bounds__(256)
void f32_to_bf16(const float* __restrict__ src, __bf16* __restrict__ dst, int n) {
    int i = blockIdx.x * 256 + threadIdx.x;
    if (i < n) dst[i] = (__bf16)src[i];
}

// ---------------------------------------------------------------------------
// Build raw (x) and temporally-subtracted (xs) bf16 streams in one pass
// ---------------------------------------------------------------------------
__global__ __launch_bounds__(256)
void build_streams(const float* __restrict__ x, __bf16* __restrict__ xbf,
                   __bf16* __restrict__ xsbf, int total) {
    int idx = blockIdx.x * 256 + threadIdx.x;
    if (idx >= total) return;
    const int per_t = N_ * C_;
    int t = (idx / per_t) % T_;
    float v = x[idx];
    xbf[idx] = (__bf16)v;
    float vs = (t >= 2) ? (v - x[idx - per_t]) : v;
    xsbf[idx] = (__bf16)vs;
}

// ---------------------------------------------------------------------------
// Generic GEMM: out[m][n] = sum_k A[m][k] * W[n][k] + bias[n]
// A row-major stride K, W row-major (out,in) -> B fragment reads contiguous K.
// Block = 256 thr = 8 waves; block tile 64(M) x 128(N); wave tile 16 x 64.
// ---------------------------------------------------------------------------
template <bool F32OUT>
__global__ __launch_bounds__(256)
void gemm_bf16(const __bf16* __restrict__ A, const __bf16* __restrict__ W,
               const float* __restrict__ bias, void* __restrict__ outp,
               int M, int N, int K) {
    const int lane = threadIdx.x & 31;
    const int w    = threadIdx.x >> 5;
    const int half = lane >> 4;
    const int r    = lane & 15;
    const int mBase = blockIdx.x * 64 + (w & 3) * 16;
    const int nBase = blockIdx.y * 128 + (w >> 2) * 64;

    v8f acc[4] = {vzero8(), vzero8(), vzero8(), vzero8()};

    int rowA = mBase + r;
    if (rowA >= M) rowA = M - 1;                     // clamp padded rows
    const __bf16* arow = A + (size_t)rowA * K;

    for (int kk = 0; kk < K; kk += 32) {
        // hide HBM/L2 latency behind the 4 WMMAs of this k-step
        __builtin_prefetch(arow + kk + 64, 0, 1);    // global_prefetch_b8
        Frag a;
        a.q[0] = *reinterpret_cast<const u32x4*>(arow + kk + 8 * half);
        a.q[1] = *reinterpret_cast<const u32x4*>(arow + kk + 16 + 8 * half);
#pragma unroll
        for (int nb = 0; nb < 4; ++nb) {
            const __bf16* wrow = W + (size_t)(nBase + nb * 16 + r) * K;
            if (nb == 0) __builtin_prefetch(wrow + kk + 64, 0, 1);
            Frag bfr;
            bfr.q[0] = *reinterpret_cast<const u32x4*>(wrow + kk + 8 * half);
            bfr.q[1] = *reinterpret_cast<const u32x4*>(wrow + kk + 16 + 8 * half);
            acc[nb] = __builtin_amdgcn_wmma_f32_16x16x32_bf16(
                false, a.v, false, bfr.v, (short)0, acc[nb], false, false);
        }
    }

#pragma unroll
    for (int nb = 0; nb < 4; ++nb) {
        const int col = nBase + nb * 16 + r;
        const float bv = bias[col];
#pragma unroll
        for (int i = 0; i < 8; ++i) {
            const int row = mBase + i + 8 * half;
            if (row < M) {
                float val = acc[nb][i] + bv;
                if (F32OUT)
                    ((float*)outp)[(size_t)row * N + col] = val;
                else
                    ((__bf16*)outp)[(size_t)row * N + col] = (__bf16)val;
            }
        }
    }
}

// ---------------------------------------------------------------------------
// Temporal attention over T=17 per (b, n, h). One wave per problem, 4 waves
// per block. VALU path (17x17 is too small for 16x16x32 tiles to pay off).
// q,k come from tqk_out (stride 2C, k at +C); v from tv_out (stride C).
// ---------------------------------------------------------------------------
__global__ __launch_bounds__(128)
void temporal_attn(const __bf16* __restrict__ tqk, const __bf16* __restrict__ tv,
                   __bf16* __restrict__ out) {
    __shared__ __attribute__((aligned(16))) __bf16 kv[4][2][T_ * D_];
    const int lane = threadIdx.x & 31;
    const int w    = threadIdx.x >> 5;
    const int gid  = blockIdx.x * 4 + w;
    const int h = gid % H_;
    const int n = (gid / H_) % N_;
    const int b = gid / (H_ * N_);

    // stage K and V tiles (17 x 64 each) into this wave's LDS slice
    for (int i = lane; i < T_ * D_; i += 32) {
        int t = i >> 6, d = i & 63;
        size_t tok = (size_t)(b * T_ + t) * N_ + n;
        kv[w][0][i] = tqk[tok * (2 * C_) + C_ + h * D_ + d];
        kv[w][1][i] = tv[tok * C_ + h * D_ + d];
    }
    asm volatile("s_wait_dscnt 0" ::: "memory");     // in-wave cross-lane LDS RAW

    if (lane < T_) {
        const __bf16* qrow =
            tqk + ((size_t)(b * T_ + lane) * N_ + n) * (2 * C_) + h * D_;
        float s[T_];
        for (int k2 = 0; k2 < T_; ++k2) s[k2] = 0.f;
        for (int d = 0; d < D_; ++d) {
            float qd = (float)qrow[d];
            for (int k2 = 0; k2 < T_; ++k2)
                s[k2] += qd * (float)kv[w][0][k2 * D_ + d];
        }
        float m = -1e30f;
        for (int k2 = 0; k2 < T_; ++k2) m = fmaxf(m, s[k2] * SCL);
        float l = 0.f;
        for (int k2 = 0; k2 < T_; ++k2) {
            s[k2] = __expf(s[k2] * SCL - m);
            l += s[k2];
        }
        const float inv = 1.f / l;
        __bf16* orow = out + ((size_t)(b * T_ + lane) * N_ + n) * C_ + h * D_;
        for (int d = 0; d < D_; ++d) {
            float o = 0.f;
            for (int k2 = 0; k2 < T_; ++k2)
                o += s[k2] * (float)kv[w][1][k2 * D_ + d];
            orow[d] = (__bf16)(o * inv);
        }
    }
}

// ---------------------------------------------------------------------------
// Spatial flash attention over N=197 per (b, h, t). One workgroup per problem,
// 13 waves, each wave owns one 16-row Q tile. Keys processed 32 at a time;
// K/V tiles staged with global_load_async_to_lds_b128 (ASYNCcnt), then
// S = Q K^T and O += P V both on v_wmma_f32_16x16x32_bf16.
// qkv layout: sqkv_out stride 3C, q at +0, k at +C, v at +2C.
// Out-of-range keys are CLAMPED (not zeroed): their scores are masked to
// -1e30 so p == 0 exactly and the clamped (finite) V rows contribute nothing.
// ---------------------------------------------------------------------------
__global__ __launch_bounds__(416)
void spatial_attn(const __bf16* __restrict__ sqkv, __bf16* __restrict__ out) {
    __shared__ __attribute__((aligned(16))) __bf16 Klds[32 * 64];
    __shared__ __attribute__((aligned(16))) __bf16 Vlds[32 * 64];
    __shared__ __attribute__((aligned(16))) __bf16 Plds[13][16 * 32];

    const int b = blockIdx.y;
    const int h = blockIdx.x / T_;
    const int t = blockIdx.x % T_;
    const int lane = threadIdx.x & 31;
    const int w    = threadIdx.x >> 5;               // 0..12 : Q tile index
    const int half = lane >> 4;
    const int r    = lane & 15;

    const size_t tokBase = (size_t)(b * T_ + t) * N_;
    const unsigned long long gbase = (unsigned long long)(uintptr_t)sqkv;

    // Preload this wave's Q fragments (16 rows x 64 d, two 32-wide k-steps)
    Frag aq[2];
    {
        int qr = w * 16 + r;
        if (qr > N_ - 1) qr = N_ - 1;
        const __bf16* qrow = sqkv + (tokBase + qr) * (3 * C_) + h * D_;
#pragma unroll
        for (int dk = 0; dk < 2; ++dk) {
            aq[dk].q[0] = *reinterpret_cast<const u32x4*>(qrow + dk * 32 + 8 * half);
            aq[dk].q[1] = *reinterpret_cast<const u32x4*>(qrow + dk * 32 + 16 + 8 * half);
        }
    }

    // LDS byte addresses for the async DMA (generic ptr low 32 bits = LDS addr)
    const unsigned kldsBase = (unsigned)(uintptr_t)(void*)&Klds[0];
    const unsigned vldsBase = (unsigned)(uintptr_t)(void*)&Vlds[0];

    float m[8], l[8];
    v8f oacc[4];
#pragma unroll
    for (int i = 0; i < 8; ++i) { m[i] = -1e30f; l[i] = 0.f; }
#pragma unroll
    for (int i = 0; i < 4; ++i) oacc[i] = vzero8();

    for (int kt = 0; kt < 7; ++kt) {                 // 7*32 = 224 >= 197 keys
        __syncthreads();                             // prior consumers done

        // ---- async stage of K,V tiles: 64 keys x 64B (d 0..31) = 256 x 16B ----
        if (threadIdx.x < 256) {
            const int c = threadIdx.x;               // 16B chunk id
            int key = kt * 32 + (c >> 2);
            if (key > N_ - 1) key = N_ - 1;          // clamp (masked later)
            const unsigned rowOff = (c & 3) * 16;    // byte offset in 64B row
            const unsigned gK =
                (unsigned)(((tokBase + key) * (3 * C_) + C_ + h * D_) * 2) + rowOff;
            const unsigned gV = gK + (unsigned)(C_ * 2);
            const unsigned ldsOff = (unsigned)c * 16u;
            unsigned kDst = kldsBase + ldsOff;
            unsigned vDst = vldsBase + ldsOff;
            asm volatile("global_load_async_to_lds_b128 %0, %1, %2"
                         :: "v"(kDst), "v"(gK), "s"(gbase) : "memory");
            asm volatile("global_load_async_to_lds_b128 %0, %1, %2"
                         :: "v"(vDst), "v"(gV), "s"(gbase) : "memory");
        }
        asm volatile("s_wait_asynccnt 0" ::: "memory");
        __syncthreads();

        // S tiles: 16 rows x 32 keys (two 16-key subtiles), K-dim = D = 64
        v8f sacc[2];
#pragma unroll
        for (int ks = 0; ks < 2; ++ks) {
            sacc[ks] = vzero8();
#pragma unroll
            for (int dk = 0; dk < 2; ++dk) {
                const __bf16* krow = &Klds[(ks * 16 + r) * 64 + dk * 32];
                Frag bk;
                bk.q[0] = *reinterpret_cast<const u32x4*>(krow + 8 * half);
                bk.q[1] = *reinterpret_cast<const u32x4*>(krow + 16 + 8 * half);
                sacc[ks] = __builtin_amdgcn_wmma_f32_16x16x32_bf16(
                    false, aq[dk].v, false, bk.v, (short)0, sacc[ks], false, false);
            }
        }

        // scale + mask (C layout: lane r = key column, VGPR i = row i+8*half)
        float p0a[8], p1a[8];
        {
            const bool v0 = (kt * 32 + r) < N_;
            const bool v1 = (kt * 32 + 16 + r) < N_;
#pragma unroll
            for (int i = 0; i < 8; ++i) {
                p0a[i] = v0 ? sacc[0][i] * SCL : -1e30f;
                p1a[i] = v1 ? sacc[1][i] * SCL : -1e30f;
            }
        }

        // online softmax per row (16-lane row groups)
#pragma unroll
        for (int i = 0; i < 8; ++i) {
            float cm = fmaxf(p0a[i], p1a[i]);
            for (int off = 8; off; off >>= 1) cm = fmaxf(cm, __shfl_xor(cm, off, 16));
            const float mn = fmaxf(m[i], cm);
            const float corr = __expf(m[i] - mn);
            const float p0 = __expf(p0a[i] - mn);
            const float p1 = __expf(p1a[i] - mn);
            float rs = p0 + p1;
            for (int off = 8; off; off >>= 1) rs += __shfl_xor(rs, off, 16);
            l[i] = l[i] * corr + rs;
            m[i] = mn;
#pragma unroll
            for (int ds = 0; ds < 4; ++ds) oacc[ds][i] = oacc[ds][i] * corr;
            Plds[w][(i + 8 * half) * 32 + r]      = (__bf16)p0;
            Plds[w][(i + 8 * half) * 32 + 16 + r] = (__bf16)p1;
        }
        asm volatile("s_wait_dscnt 0" ::: "memory"); // in-wave P transpose RAW

        // A fragment of P (16 rows x 32 keys)
        Frag ap;
        {
            const __bf16* prow = &Plds[w][r * 32];
            ap.q[0] = *reinterpret_cast<const u32x4*>(prow + 8 * half);
            ap.q[1] = *reinterpret_cast<const u32x4*>(prow + 16 + 8 * half);
        }

        // O += P V  (4 d-subtiles of 16)
#pragma unroll
        for (int ds = 0; ds < 4; ++ds) {
            Frag bv;
#pragma unroll
            for (int j = 0; j < 8; ++j) {
                bv.e[j]     = Vlds[(8 * half + j) * 64 + ds * 16 + r];
                bv.e[8 + j] = Vlds[(16 + 8 * half + j) * 64 + ds * 16 + r];
            }
            oacc[ds] = __builtin_amdgcn_wmma_f32_16x16x32_bf16(
                false, ap.v, false, bv.v, (short)0, oacc[ds], false, false);
        }
    }

    // normalize and store
#pragma unroll
    for (int i = 0; i < 8; ++i) {
        const int qr = w * 16 + i + 8 * half;
        if (qr < N_) {
            const float inv = 1.f / l[i];
            __bf16* orow = out + (tokBase + qr) * C_ + h * D_;
#pragma unroll
            for (int ds = 0; ds < 4; ++ds)
                orow[ds * 16 + r] = (__bf16)(oacc[ds][i] * inv);
        }
    }
}

// ---------------------------------------------------------------------------
// Host-side orchestration
// ---------------------------------------------------------------------------
extern "C" void kernel_launch(void* const* d_in, const int* in_sizes, int n_in,
                              void* d_out, int out_size, void* d_ws, size_t ws_size,
                              hipStream_t stream) {
    (void)in_sizes; (void)n_in; (void)out_size; (void)ws_size;

    const float* x       = (const float*)d_in[0];
    const float* sqkv_w  = (const float*)d_in[1];
    const float* sqkv_b  = (const float*)d_in[2];
    const float* tqk_w   = (const float*)d_in[3];
    const float* tqk_b   = (const float*)d_in[4];
    const float* tv_w    = (const float*)d_in[5];
    const float* tv_b    = (const float*)d_in[6];
    const float* sproj_w = (const float*)d_in[7];
    const float* sproj_b = (const float*)d_in[8];
    const float* tproj_w = (const float*)d_in[9];
    const float* tproj_b = (const float*)d_in[10];

    char* ws = (char*)d_ws;
    auto alloc = [&](size_t bytes) {
        char* p = ws;
        ws += (bytes + 255) & ~(size_t)255;
        return p;
    };

    const size_t E = (size_t)TOK * C_;               // 20,576,256 elements

    __bf16* w_sqkv = (__bf16*)alloc((size_t)3 * C_ * C_ * 2);
    __bf16* w_tqk  = (__bf16*)alloc((size_t)2 * C_ * C_ * 2);
    __bf16* w_tv   = (__bf16*)alloc((size_t)C_ * C_ * 2);
    __bf16* w_sp   = (__bf16*)alloc((size_t)C_ * C_ * 2);
    __bf16* w_tp   = (__bf16*)alloc((size_t)C_ * C_ * 2);
    // region1: xbf, later reused as tproj_out
    __bf16* xbf    = (__bf16*)alloc(E * 2);
    // region2: xsbf -> t_attn_out -> s_attn_out (strictly sequential lifetimes)
    __bf16* xsbf   = (__bf16*)alloc(E * 2);
    // region3: tqk_out(2E) + tv_out(E), later overwritten by sqkv_out(3E)
    __bf16* big    = (__bf16*)alloc(3 * E * 2);

    __bf16* tqkOut = big;
    __bf16* tvOut  = big + 2 * E;
    __bf16* tattn  = xsbf;
    __bf16* tprojO = xbf;
    __bf16* sqkvO  = big;
    __bf16* sattn  = xsbf;
    float*  outF   = (float*)d_out;

    // weight conversion
    f32_to_bf16<<<(3 * C_ * C_ + 255) / 256, 256, 0, stream>>>(sqkv_w, w_sqkv, 3 * C_ * C_);
    f32_to_bf16<<<(2 * C_ * C_ + 255) / 256, 256, 0, stream>>>(tqk_w, w_tqk, 2 * C_ * C_);
    f32_to_bf16<<<(C_ * C_ + 255) / 256, 256, 0, stream>>>(tv_w, w_tv, C_ * C_);
    f32_to_bf16<<<(C_ * C_ + 255) / 256, 256, 0, stream>>>(sproj_w, w_sp, C_ * C_);
    f32_to_bf16<<<(C_ * C_ + 255) / 256, 256, 0, stream>>>(tproj_w, w_tp, C_ * C_);

    build_streams<<<((int)E + 255) / 256, 256, 0, stream>>>(x, xbf, xsbf, (int)E);

    const int MB = (TOK + 63) / 64;                  // 419 M-blocks
    // temporal q,k from subtracted stream
    gemm_bf16<false><<<dim3(MB, (2 * C_) / 128), 256, 0, stream>>>(
        xsbf, w_tqk, tqk_b, tqkOut, TOK, 2 * C_, C_);
    // temporal v from raw stream
    gemm_bf16<false><<<dim3(MB, C_ / 128), 256, 0, stream>>>(
        xbf, w_tv, tv_b, tvOut, TOK, C_, C_);

    temporal_attn<<<(B_ * N_ * H_) / 4, 128, 0, stream>>>(tqkOut, tvOut, tattn);

    gemm_bf16<false><<<dim3(MB, C_ / 128), 256, 0, stream>>>(
        tattn, w_tp, tproj_b, tprojO, TOK, C_, C_);
    gemm_bf16<false><<<dim3(MB, (3 * C_) / 128), 256, 0, stream>>>(
        tprojO, w_sqkv, sqkv_b, sqkvO, TOK, 3 * C_, C_);

    spatial_attn<<<dim3(H_ * T_, B_), 416, 0, stream>>>(sqkvO, sattn);

    gemm_bf16<true><<<dim3(MB, C_ / 128), 256, 0, stream>>>(
        sattn, w_sp, sproj_b, (void*)outF, TOK, C_, C_);
}
